// FunctionG_17471926960640
// MI455X (gfx1250) — compile-verified, matrix operand order checked by reading, not run on hardware
//
#include <hip/hip_runtime.h>
#include <math.h>

// Model dims
#define B_    32
#define C_    128
#define L_    64
#define H_    16
#define HID_  128
#define DIN_  192
#define NTR_  544      // 32 cur + 512 hist traces
#define NC_   4096     // B*C candidates

typedef __attribute__((ext_vector_type(2)))  float  v2f;
typedef __attribute__((ext_vector_type(8)))  float  v8f;
typedef __attribute__((ext_vector_type(8)))  __bf16 v8bf;
typedef __attribute__((ext_vector_type(16))) __bf16 v16bf;

// ---------------------------------------------------------------------------
// helpers
// ---------------------------------------------------------------------------
__device__ __forceinline__ __bf16 f2bf(float f) {
  unsigned u = __builtin_bit_cast(unsigned, f);
  u += 0x7FFFu + ((u >> 16) & 1u);           // round-to-nearest-even
  unsigned short s = (unsigned short)(u >> 16);
  return __builtin_bit_cast(__bf16, s);
}

__device__ __forceinline__ v8f wmma_bf(v16bf a, v16bf b, v8f c) {
  return __builtin_amdgcn_wmma_f32_16x16x32_bf16(false, a, false, b, (short)0, c,
                                                 false, false);
}
__device__ __forceinline__ v8f wmma4(v2f a, v2f b, v8f c) {
  return __builtin_amdgcn_wmma_f32_16x16x4_f32(false, a, false, b, (short)0, c,
                                               false, false);
}

// 16-bit WMMA fragment (A row m=lane&15 / B col n=lane&15) from a row-major
// bf16 row pointer: two contiguous 16-byte chunks:
//   elems 0..7  = K[kb    + g*8 ..],   elems 8..15 = K[kb+16 + g*8 ..]
__device__ __forceinline__ v16bf load_frag(const __bf16* rowp, int kb, int lane) {
  const int g8 = (lane >> 4) << 3;  // 0 or 8
  v8bf lo = *(const v8bf*)(rowp + kb + g8);
  v8bf hi = *(const v8bf*)(rowp + kb + 16 + g8);
  return __builtin_shufflevector(lo, hi, 0, 1, 2, 3, 4, 5, 6, 7, 8, 9, 10, 11,
                                 12, 13, 14, 15);
}

// D tile (16x16 f32): lane n = lane&15, row m = v + 8*(lane>=16)
__device__ __forceinline__ void store_tile(float* __restrict__ out, int ldo,
                                           v8f acc, int lane) {
  const int n  = lane & 15;
  const int mb = (lane >> 4) << 3;
#pragma unroll
  for (int v = 0; v < 8; ++v) out[(mb + v) * ldo + n] = acc[v];
}

__device__ __forceinline__ float sigm(float x) { return 1.0f / (1.0f + __expf(-x)); }

__device__ __forceinline__ float wred_sum(float v) {
#pragma unroll
  for (int o = 16; o > 0; o >>= 1) v += __shfl_xor(v, o, 32);
  return v;
}
__device__ __forceinline__ float wred_max(float v) {
#pragma unroll
  for (int o = 16; o > 0; o >>= 1) v = fmaxf(v, __shfl_xor(v, o, 32));
  return v;
}

// ---------------------------------------------------------------------------
// K0: f32 -> bf16 conversion (weights)
// ---------------------------------------------------------------------------
__global__ __launch_bounds__(256) void f32_to_bf16(const float* __restrict__ src,
                                                   __bf16* __restrict__ dst,
                                                   int n) {
  const int i = blockIdx.x * 256 + threadIdx.x;
  if (i < n) dst[i] = f2bf(src[i]);
}

// ---------------------------------------------------------------------------
// GRU kernels.  768 threads = 24 waves:
//   waves  0..15 : r/z tile nt=w :  grz = x@Wih^T + h@Whh^T + b_ih + b_hh
//   waves 16..23 : n-gate tile j  :  gxn = x@Wihn^T + b_ihn ; ghn = h@Whhn^T + b_hhn
// Weight B-fragments (bf16) live in VGPRs across the whole time loop:
//   per wave: 6 frags (K=192, x) + 4 frags (K=128, h) = 80 VGPRs.
// Steady state per wave per step: 20x ds_load_b128 + 10x v_wmma_..._bf16.
// ---------------------------------------------------------------------------
__global__ __launch_bounds__(768) void gru_main(
    const int* __restrict__ uid, const int* __restrict__ cur_loc,
    const int* __restrict__ cur_wd, const int* __restrict__ cur_hr,
    const int* __restrict__ hist_loc, const int* __restrict__ hist_wd,
    const int* __restrict__ hist_hr, const float* __restrict__ uid_emb,
    const float* __restrict__ loc_emb, const float* __restrict__ wd_emb,
    const float* __restrict__ hr_emb, const __bf16* __restrict__ Wih_bf,
    const __bf16* __restrict__ Whh_bf, const float* __restrict__ b_ih,
    const float* __restrict__ b_hh, float* __restrict__ hs) {
  __shared__ __bf16 sx[16 * DIN_];
  __shared__ __bf16 shb[16 * HID_];
  __shared__ float  sh[16 * HID_];
  __shared__ float  grz[16 * 256];
  __shared__ float  gxn[16 * HID_];
  __shared__ float  ghn[16 * HID_];

  const int tid = threadIdx.x, lane = tid & 31, w = tid >> 5, ln = lane & 15;
  const int tr0 = blockIdx.x * 16;
  const bool is_rz = (w < 16);

  // ---- preload weight fragments + biases (persistent across time loop) ----
  const int row = is_rz ? (w * 16 + ln) : (256 + (w - 16) * 16 + ln);
  v16bf Bx[6], Bh[4];
#pragma unroll
  for (int i = 0; i < 6; ++i) Bx[i] = load_frag(Wih_bf + row * DIN_, i * 32, lane);
#pragma unroll
  for (int i = 0; i < 4; ++i) Bh[i] = load_frag(Whh_bf + row * HID_, i * 32, lane);
  const int bn = is_rz ? (w * 16 + ln) : (256 + (w - 16) * 16 + ln);
  const float bias_x = b_ih[bn];
  const float bias_h = b_hh[bn];

  for (int i = tid; i < 16 * HID_; i += 768) { sh[i] = 0.0f; shb[i] = f2bf(0.0f); }

  for (int t = 0; t < L_; ++t) {
    __syncthreads();
    // gather embeddings -> x (16 x 192, bf16)
    for (int i = tid; i < 16 * DIN_; i += 768) {
      const int m = i / DIN_, d = i - m * DIN_;
      const int tr = tr0 + m;
      float v;
      if (d < 32) {
        const int u = (tr < 32) ? uid[tr] : uid[(tr - 32) >> 4];
        v = uid_emb[u * 32 + d];
      } else if (d < 160) {
        const int loc = (tr < 32) ? cur_loc[tr * L_ + t] : hist_loc[(tr - 32) * L_ + t];
        v = loc_emb[loc * 128 + (d - 32)];
      } else if (d < 176) {
        const int wd = (tr < 32) ? cur_wd[tr * L_ + t] : hist_wd[(tr - 32) * L_ + t];
        v = wd_emb[wd * 16 + (d - 160)];
      } else {
        const int hr = (tr < 32) ? cur_hr[tr * L_ + t] : hist_hr[(tr - 32) * L_ + t];
        v = hr_emb[hr * 16 + (d - 176)];
      }
      sx[i] = f2bf(v);
    }
    __syncthreads();
    // ---- gate GEMMs (WMMA bf16, weights already in VGPRs) ----
    if (is_rz) {
      v8f acc;
#pragma unroll
      for (int v = 0; v < 8; ++v) acc[v] = bias_x + bias_h;
#pragma unroll
      for (int i = 0; i < 6; ++i)
        acc = wmma_bf(load_frag(sx + ln * DIN_, i * 32, lane), Bx[i], acc);
#pragma unroll
      for (int i = 0; i < 4; ++i)
        acc = wmma_bf(load_frag(shb + ln * HID_, i * 32, lane), Bh[i], acc);
      store_tile(grz + w * 16, 256, acc, lane);
    } else {
      const int c0 = (w - 16) * 16;
      v8f ax;
#pragma unroll
      for (int v = 0; v < 8; ++v) ax[v] = bias_x;
#pragma unroll
      for (int i = 0; i < 6; ++i)
        ax = wmma_bf(load_frag(sx + ln * DIN_, i * 32, lane), Bx[i], ax);
      store_tile(gxn + c0, HID_, ax, lane);
      v8f ah;
#pragma unroll
      for (int v = 0; v < 8; ++v) ah[v] = bias_h;
#pragma unroll
      for (int i = 0; i < 4; ++i)
        ah = wmma_bf(load_frag(shb + ln * HID_, i * 32, lane), Bh[i], ah);
      store_tile(ghn + c0, HID_, ah, lane);
    }
    __syncthreads();
    // ---- gate nonlinearity + state update ----
    if (tid < 512) {
#pragma unroll
      for (int e = 0; e < 4; ++e) {
        const int flat = tid * 4 + e;
        const int m = flat >> 7, k = flat & 127;
        const float r = sigm(grz[m * 256 + k]);
        const float z = sigm(grz[m * 256 + 128 + k]);
        const float n = tanhf(gxn[m * HID_ + k] + r * ghn[m * HID_ + k]);
        const float hn = (1.0f - z) * n + z * sh[m * HID_ + k];
        sh[m * HID_ + k]  = hn;
        shb[m * HID_ + k] = f2bf(hn);
        hs[(size_t)(tr0 + m) * L_ * HID_ + (size_t)t * HID_ + k] = hn;
      }
    }
  }
}

// one GRU step for all 4096 candidates (shared h_63 per batch), 16/block
__global__ __launch_bounds__(768) void gru_cand(
    const int* __restrict__ uid, const int* __restrict__ cand,
    const int* __restrict__ cur_wd, const int* __restrict__ cur_hr,
    const float* __restrict__ uid_emb, const float* __restrict__ loc_emb,
    const float* __restrict__ wd_emb, const float* __restrict__ hr_emb,
    const __bf16* __restrict__ Wih_bf, const __bf16* __restrict__ Whh_bf,
    const float* __restrict__ b_ih, const float* __restrict__ b_hh,
    const float* __restrict__ hs, float* __restrict__ hcand) {
  __shared__ __bf16 sx[16 * DIN_];
  __shared__ __bf16 shb[16 * HID_];
  __shared__ float  sh[16 * HID_];
  __shared__ float  grz[16 * 256];
  __shared__ float  gxn[16 * HID_];
  __shared__ float  ghn[16 * HID_];

  const int tid = threadIdx.x, lane = tid & 31, w = tid >> 5, ln = lane & 15;
  const int bc0 = blockIdx.x * 16;
  const int b   = bc0 >> 7;
  const bool is_rz = (w < 16);

  const int row = is_rz ? (w * 16 + ln) : (256 + (w - 16) * 16 + ln);
  v16bf Bx[6], Bh[4];
#pragma unroll
  for (int i = 0; i < 6; ++i) Bx[i] = load_frag(Wih_bf + row * DIN_, i * 32, lane);
#pragma unroll
  for (int i = 0; i < 4; ++i) Bh[i] = load_frag(Whh_bf + row * HID_, i * 32, lane);
  const float bias_x = b_ih[row];
  const float bias_h = b_hh[row];

  const int wdv = cur_wd[b * L_ + (L_ - 1)];
  const int hrv = cur_hr[b * L_ + (L_ - 1)];
  const int uv  = uid[b];

  for (int i = tid; i < 16 * HID_; i += 768) {
    const int k = i & 127;
    const float hv = hs[(size_t)(b * L_ + (L_ - 1)) * HID_ + k];  // shared h_63
    sh[i] = hv;
    shb[i] = f2bf(hv);
  }
  for (int i = tid; i < 16 * DIN_; i += 768) {
    const int m = i / DIN_, d = i - m * DIN_;
    float v;
    if (d < 32)       v = uid_emb[uv * 32 + d];
    else if (d < 160) v = loc_emb[cand[bc0 + m] * 128 + (d - 32)];
    else if (d < 176) v = wd_emb[wdv * 16 + (d - 160)];
    else              v = hr_emb[hrv * 16 + (d - 176)];
    sx[i] = f2bf(v);
  }
  __syncthreads();
  if (is_rz) {
    v8f acc;
#pragma unroll
    for (int v = 0; v < 8; ++v) acc[v] = bias_x + bias_h;
#pragma unroll
    for (int i = 0; i < 6; ++i)
      acc = wmma_bf(load_frag(sx + ln * DIN_, i * 32, lane), Bx[i], acc);
#pragma unroll
    for (int i = 0; i < 4; ++i)
      acc = wmma_bf(load_frag(shb + ln * HID_, i * 32, lane), Bh[i], acc);
    store_tile(grz + w * 16, 256, acc, lane);
  } else {
    const int c0 = (w - 16) * 16;
    v8f ax;
#pragma unroll
    for (int v = 0; v < 8; ++v) ax[v] = bias_x;
#pragma unroll
    for (int i = 0; i < 6; ++i)
      ax = wmma_bf(load_frag(sx + ln * DIN_, i * 32, lane), Bx[i], ax);
    store_tile(gxn + c0, HID_, ax, lane);
    v8f ah;
#pragma unroll
    for (int v = 0; v < 8; ++v) ah[v] = bias_h;
#pragma unroll
    for (int i = 0; i < 4; ++i)
      ah = wmma_bf(load_frag(shb + ln * HID_, i * 32, lane), Bh[i], ah);
    store_tile(ghn + c0, HID_, ah, lane);
  }
  __syncthreads();
  if (tid < 512) {
#pragma unroll
    for (int e = 0; e < 4; ++e) {
      const int flat = tid * 4 + e;
      const int m = flat >> 7, k = flat & 127;
      const float r = sigm(grz[m * 256 + k]);
      const float z = sigm(grz[m * 256 + 128 + k]);
      const float n = tanhf(gxn[m * HID_ + k] + r * ghn[m * HID_ + k]);
      hcand[(size_t)(bc0 + m) * HID_ + k] =
          (1.0f - z) * n + z * sh[m * HID_ + k];
    }
  }
}

// ---------------------------------------------------------------------------
// K3: Out = relu(A @ W^T), K = N = 128, f32 WMMA.  grid.x = M/16,
// 8 waves = 8 n-tiles.  Dual accumulators + bounded unroll: no spills,
// loads overlap the two independent accumulate chains.
// ---------------------------------------------------------------------------
__global__ __launch_bounds__(256) void gemm128_relu(
    const float* __restrict__ A, int strideA, const float* __restrict__ W,
    float* __restrict__ Out, int strideOut) {
  const int lane = threadIdx.x & 31;
  const int w    = threadIdx.x >> 5;
  const int m0   = blockIdx.x * 16;
  const int n0   = w * 16;
  const int rm   = lane & 15;
  const int off  = (lane >> 4) << 1;
  const float* ap = A + (size_t)(m0 + rm) * strideA + off;
  const float* wp = W + (size_t)(n0 + rm) * HID_ + off;
  v8f acc0, acc1;
#pragma unroll
  for (int v = 0; v < 8; ++v) { acc0[v] = 0.0f; acc1[v] = 0.0f; }
#pragma unroll 4
  for (int kb = 0; kb < HID_; kb += 8) {
    acc0 = wmma4(*(const v2f*)(ap + kb), *(const v2f*)(wp + kb), acc0);
    acc1 = wmma4(*(const v2f*)(ap + kb + 4), *(const v2f*)(wp + kb + 4), acc1);
  }
  const int n  = lane & 15;
  const int mb = (lane >> 4) << 3;
#pragma unroll
  for (int v = 0; v < 8; ++v) {
    const float x = acc0[v] + acc1[v];
    Out[(size_t)(m0 + mb + v) * strideOut + n0 + n] = x > 0.0f ? x : 0.0f;
  }
}

// ---------------------------------------------------------------------------
// K4: intra attention for candidates (wave/candidate, T = 65 = 64 shared + own)
// ---------------------------------------------------------------------------
__global__ __launch_bounds__(256) void attn_cand(
    const float* __restrict__ qh_c, const float* __restrict__ kh_all,
    const float* __restrict__ kh_c, const float* __restrict__ hs,
    const float* __restrict__ hcand, const float* __restrict__ out_w,
    float* __restrict__ out) {
  const int g = (int)((blockIdx.x * blockDim.x + threadIdx.x) >> 5);
  const int lane = threadIdx.x & 31;
  if (g >= NC_) return;
  const int b = g >> 7;
  const float* khb = kh_all + (size_t)b * L_ * HID_;
  const float* vb  = hs + (size_t)b * L_ * HID_;
  float q[4], ow[4];
#pragma unroll
  for (int j = 0; j < 4; ++j) {
    q[j]  = qh_c[(size_t)g * HID_ + lane + 32 * j];
    ow[j] = out_w[lane + 32 * j];
  }
  float s0 = 0.f, s1 = 0.f, s2 = -1e30f;
  for (int t = 0; t < L_ + 1; ++t) {
    const float* kh = (t < L_) ? (khb + (size_t)t * HID_) : (kh_c + (size_t)g * HID_);
    float p = 0.f;
#pragma unroll
    for (int j = 0; j < 4; ++j) p += tanhf(q[j] + kh[lane + 32 * j]) * ow[j];
    p = wred_sum(p);
    const int slot = t >> 5, pos = t & 31;
    if (pos == lane) { if (slot == 0) s0 = p; else if (slot == 1) s1 = p; else s2 = p; }
  }
  const float m  = wred_max(fmaxf(fmaxf(s0, s1), s2));
  const float e0 = __expf(s0 - m), e1 = __expf(s1 - m);
  const float e2 = (lane == 0) ? __expf(s2 - m) : 0.f;
  const float inv = 1.0f / wred_sum(e0 + e1 + e2);
  float acc[4] = {0.f, 0.f, 0.f, 0.f};
  for (int t = 0; t < L_ + 1; ++t) {
    const int slot = t >> 5, pos = t & 31;
    const float at = __shfl((slot == 0) ? e0 : (slot == 1) ? e1 : e2, pos, 32) * inv;
    const float* vv = (t < L_) ? (vb + (size_t)t * HID_) : (hcand + (size_t)g * HID_);
#pragma unroll
    for (int j = 0; j < 4; ++j) acc[j] += at * vv[lane + 32 * j];
  }
#pragma unroll
  for (int j = 0; j < 4; ++j) out[(size_t)g * HID_ + lane + 32 * j] = acc[j];
}

// ---------------------------------------------------------------------------
// K5: intra attention for history traces (wave/trace, T = 64)
// ---------------------------------------------------------------------------
__global__ __launch_bounds__(256) void attn_hist(
    const float* __restrict__ qh_h, const float* __restrict__ kh_all,
    const float* __restrict__ hs, const float* __restrict__ out_w,
    float* __restrict__ out) {
  const int g = (int)((blockIdx.x * blockDim.x + threadIdx.x) >> 5);
  const int lane = threadIdx.x & 31;
  if (g >= B_ * H_) return;
  const float* khb = kh_all + (size_t)(32 + g) * L_ * HID_;
  const float* vb  = hs + (size_t)(32 + g) * L_ * HID_;
  float q[4], ow[4];
#pragma unroll
  for (int j = 0; j < 4; ++j) {
    q[j]  = qh_h[(size_t)g * HID_ + lane + 32 * j];
    ow[j] = out_w[lane + 32 * j];
  }
  float s0 = 0.f, s1 = 0.f;
  for (int t = 0; t < L_; ++t) {
    const float* kh = khb + (size_t)t * HID_;
    float p = 0.f;
#pragma unroll
    for (int j = 0; j < 4; ++j) p += tanhf(q[j] + kh[lane + 32 * j]) * ow[j];
    p = wred_sum(p);
    if ((t & 31) == lane) { if ((t >> 5) == 0) s0 = p; else s1 = p; }
  }
  const float m  = wred_max(fmaxf(s0, s1));
  const float e0 = __expf(s0 - m), e1 = __expf(s1 - m);
  const float inv = 1.0f / wred_sum(e0 + e1);
  float acc[4] = {0.f, 0.f, 0.f, 0.f};
  for (int t = 0; t < L_; ++t) {
    const float at = __shfl(((t >> 5) == 0) ? e0 : e1, t & 31, 32) * inv;
    const float* vv = vb + (size_t)t * HID_;
#pragma unroll
    for (int j = 0; j < 4; ++j) acc[j] += at * vv[lane + 32 * j];
  }
#pragma unroll
  for (int j = 0; j < 4; ++j) out[(size_t)g * HID_ + lane + 32 * j] = acc[j];
}

// ---------------------------------------------------------------------------
// K6: inter attention + moving_state + score (wave/candidate, T = 16)
// ---------------------------------------------------------------------------
__global__ __launch_bounds__(256) void inter_attn(
    const float* __restrict__ qh2, const float* __restrict__ kh2,
    const float* __restrict__ hist_hid, const float* __restrict__ out_w,
    const float* __restrict__ score_w, float* __restrict__ ms_out,
    float* __restrict__ score) {
  const int g = (int)((blockIdx.x * blockDim.x + threadIdx.x) >> 5);
  const int lane = threadIdx.x & 31;
  if (g >= NC_) return;
  const int b = g >> 7;
  float q[4], ow[4], sw[4];
#pragma unroll
  for (int j = 0; j < 4; ++j) {
    q[j]  = qh2[(size_t)g * HID_ + lane + 32 * j];
    ow[j] = out_w[lane + 32 * j];
    sw[j] = score_w[lane + 32 * j];
  }
  float sv = -1e30f;
  for (int h = 0; h < H_; ++h) {
    const float* kh = kh2 + (size_t)(b * H_ + h) * HID_;
    float p = 0.f;
#pragma unroll
    for (int j = 0; j < 4; ++j) p += tanhf(q[j] + kh[lane + 32 * j]) * ow[j];
    p = wred_sum(p);
    if (lane == h) sv = p;
  }
  const float m = wred_max(sv);
  const float e = (lane < H_) ? __expf(sv - m) : 0.f;
  const float inv = 1.0f / wred_sum(e);
  float acc[4] = {0.f, 0.f, 0.f, 0.f};
  for (int h = 0; h < H_; ++h) {
    const float at = __shfl(e, h, 32) * inv;
    const float* vv = hist_hid + (size_t)(b * H_ + h) * HID_;
#pragma unroll
    for (int j = 0; j < 4; ++j) acc[j] += at * vv[lane + 32 * j];
  }
  float sc = 0.f;
#pragma unroll
  for (int j = 0; j < 4; ++j) {
    ms_out[(size_t)g * HID_ + lane + 32 * j] = acc[j];
    sc += acc[j] * sw[j];
  }
  sc = wred_sum(sc);
  if (lane == 0) score[g] = sc;
}

// ---------------------------------------------------------------------------
// K7: softmax over C=128 candidates per batch
// ---------------------------------------------------------------------------
__global__ __launch_bounds__(128) void softmax_c(const float* __restrict__ score,
                                                 float* __restrict__ out) {
  __shared__ float buf[128];
  const int b = blockIdx.x, tid = threadIdx.x;
  const float v = score[b * C_ + tid];
  buf[tid] = v;
  __syncthreads();
  for (int o = 64; o > 0; o >>= 1) {
    if (tid < o) buf[tid] = fmaxf(buf[tid], buf[tid + o]);
    __syncthreads();
  }
  const float m = buf[0];
  __syncthreads();
  const float e = __expf(v - m);
  buf[tid] = e;
  __syncthreads();
  for (int o = 64; o > 0; o >>= 1) {
    if (tid < o) buf[tid] += buf[tid + o];
    __syncthreads();
  }
  out[b * C_ + tid] = e / buf[0];
}

// ---------------------------------------------------------------------------
extern "C" void kernel_launch(void* const* d_in, const int* in_sizes, int n_in,
                              void* d_out, int out_size, void* d_ws,
                              size_t ws_size, hipStream_t stream) {
  (void)in_sizes; (void)n_in; (void)out_size; (void)ws_size;
  const int* uid      = (const int*)d_in[0];
  const int* cur_loc  = (const int*)d_in[1];
  const int* cur_wd   = (const int*)d_in[2];
  const int* cur_hr   = (const int*)d_in[3];
  const int* hist_loc = (const int*)d_in[4];
  const int* hist_wd  = (const int*)d_in[5];
  const int* hist_hr  = (const int*)d_in[6];
  const int* cand     = (const int*)d_in[7];
  const float* uid_emb  = (const float*)d_in[8];
  const float* loc_emb  = (const float*)d_in[9];
  const float* wd_emb   = (const float*)d_in[10];
  const float* hr_emb   = (const float*)d_in[11];
  const float* W_ih     = (const float*)d_in[12];
  const float* W_hh     = (const float*)d_in[13];
  const float* b_ih     = (const float*)d_in[14];
  const float* b_hh     = (const float*)d_in[15];
  const float* intra_w1 = (const float*)d_in[16];
  const float* intra_w2 = (const float*)d_in[17];
  const float* intra_out= (const float*)d_in[18];
  const float* inter_w1 = (const float*)d_in[19];
  const float* inter_w2 = (const float*)d_in[20];
  const float* inter_out= (const float*)d_in[21];
  const float* score_w  = (const float*)d_in[22];
  float* out = (float*)d_out;
  float* ws  = (float*)d_ws;

  // workspace layout (floats)
  size_t o = 0;
  float* hs       = ws + o; o += (size_t)NTR_ * L_ * HID_;   // 4.45M
  float* kh_all   = ws + o; o += (size_t)NTR_ * L_ * HID_;   // 4.45M
  float* hcand    = ws + o; o += (size_t)NC_ * HID_;
  float* kh_c     = ws + o; o += (size_t)NC_ * HID_;
  float* qh_c     = ws + o; o += (size_t)NC_ * HID_;
  float* qh_h     = ws + o; o += (size_t)B_ * H_ * HID_;
  float* cand_hid = ws + o; o += (size_t)NC_ * HID_;
  float* hist_hid = ws + o; o += (size_t)B_ * H_ * HID_;
  float* qh2      = ws + o; o += (size_t)NC_ * HID_;
  float* kh2      = ws + o; o += (size_t)B_ * H_ * HID_;
  float* score    = ws + o; o += (size_t)NC_;
  __bf16* wih_bf  = (__bf16*)(ws + o); o += (size_t)(384 * DIN_) / 2;
  __bf16* whh_bf  = (__bf16*)(ws + o); o += (size_t)(384 * HID_) / 2;

  // 0) weights -> bf16
  f32_to_bf16<<<(384 * DIN_ + 255) / 256, 256, 0, stream>>>(W_ih, wih_bf, 384 * DIN_);
  f32_to_bf16<<<(384 * HID_ + 255) / 256, 256, 0, stream>>>(W_hh, whh_bf, 384 * HID_);
  // 1) GRU over shared-prefix + history traces
  gru_main<<<NTR_ / 16, 768, 0, stream>>>(uid, cur_loc, cur_wd, cur_hr,
                                          hist_loc, hist_wd, hist_hr, uid_emb,
                                          loc_emb, wd_emb, hr_emb, wih_bf,
                                          whh_bf, b_ih, b_hh, hs);
  // 2) one GRU step per candidate
  gru_cand<<<NC_ / 16, 768, 0, stream>>>(uid, cand, cur_wd, cur_hr, uid_emb,
                                         loc_emb, wd_emb, hr_emb, wih_bf,
                                         whh_bf, b_ih, b_hh, hs, hcand);
  // 3) intra-attention projections (WMMA GEMM + ReLU)
  gemm128_relu<<<(NTR_ * L_) / 16, 256, 0, stream>>>(hs, HID_, intra_w2, kh_all, HID_);
  gemm128_relu<<<NC_ / 16, 256, 0, stream>>>(hcand, HID_, intra_w2, kh_c, HID_);
  gemm128_relu<<<NC_ / 16, 256, 0, stream>>>(hcand, HID_, intra_w1, qh_c, HID_);
  gemm128_relu<<<(B_ * H_) / 16, 256, 0, stream>>>(
      hs + (size_t)(32 * L_ + (L_ - 1)) * HID_, L_ * HID_, intra_w1, qh_h, HID_);
  // 4) intra attention reductions
  attn_cand<<<NC_ / 8, 256, 0, stream>>>(qh_c, kh_all, kh_c, hs, hcand,
                                         intra_out, cand_hid);
  attn_hist<<<(B_ * H_) / 8, 256, 0, stream>>>(qh_h, kh_all, hs, intra_out,
                                               hist_hid);
  // 5) inter-attention projections
  gemm128_relu<<<NC_ / 16, 256, 0, stream>>>(cand_hid, HID_, inter_w1, qh2, HID_);
  gemm128_relu<<<(B_ * H_) / 16, 256, 0, stream>>>(hist_hid, HID_, inter_w2, kh2, HID_);
  // 6) inter attention -> moving_state (output) + score
  inter_attn<<<NC_ / 8, 256, 0, stream>>>(qh2, kh2, hist_hid, inter_out,
                                          score_w, out + NC_, score);
  // 7) softmax over candidates -> probs (output)
  softmax_c<<<B_, 128, 0, stream>>>(score, out);
}
